// FeedForwardQuantum_65481071405846
// MI455X (gfx1250) — matrix operand order repaired and checked by reading, not compiled, not run on hardware
//
#include <hip/hip_runtime.h>

typedef _Float16 h16;
typedef __attribute__((ext_vector_type(16))) _Float16 v16h;
typedef __attribute__((ext_vector_type(8)))  _Float16 v8h;
typedef __attribute__((ext_vector_type(8)))  float    v8f;

#define EMBED 1024
#define FFN   4096
#define NQ    8
#define NTOK  16384   // B*S = 8*2048

#define BM 128        // tokens per block
#define BN 128        // embed cols per block
#define KC 64         // FFN chunk staged in LDS
#define HSTR 72       // padded f16 stride of LDS h tile (9*16B -> conflict-free)

// ---------------------------------------------------------------------------
// Prep: convert w2 [EMBED x FFN] f32 -> f16 in workspace (8 MiB needed).
// ---------------------------------------------------------------------------
__global__ __launch_bounds__(256) void w2_cvt_kernel(const float* __restrict__ w2,
                                                     h16* __restrict__ w2h) {
  int i = (blockIdx.x * 256 + threadIdx.x) * 8;
  float4 a = *(const float4*)(w2 + i);
  float4 b = *(const float4*)(w2 + i + 4);
  v8h o;
  o[0] = (h16)a.x; o[1] = (h16)a.y; o[2] = (h16)a.z; o[3] = (h16)a.w;
  o[4] = (h16)b.x; o[5] = (h16)b.y; o[6] = (h16)b.z; o[7] = (h16)b.w;
  *(v8h*)(w2h + i) = o;
}

// ---------------------------------------------------------------------------
// Fused: q = cos(x+theta)  ->  h = relu(q@w1^T + b1) (LDS, chunked over FFN)
//        ->  out = h @ w2^T + b2,  all matmuls via v_wmma_f32_16x16x32_f16.
// b1 rides the stage-1 WMMA C operand; b2 is pre-loaded into the accumulators.
// w2 B-fragments are loaded at the top of each chunk so their latency is
// hidden under stage-1 WMMA work + the barrier.
// ---------------------------------------------------------------------------
__global__ __launch_bounds__(256) void ffq_fused_kernel(
    const float* __restrict__ x,  const float* __restrict__ theta,
    const float* __restrict__ w1, const float* __restrict__ b1,
    const h16*  __restrict__ w2h, const float* __restrict__ b2,
    float* __restrict__ out)
{
  __shared__ h16 hlds[2 * BM * HSTR];    // double-buffered: 2 x 18 KB

  const int tid  = threadIdx.x;
  const int wave = tid >> 5;             // 8 waves (wave32)
  const int lane = tid & 31;
  const int r    = lane & 15;
  const int hi   = lane >> 4;

  const int mb = blockIdx.x >> 3;        // 128 token blocks
  const int nb = blockIdx.x & 7;         // 8 embed blocks
  const int M0 = mb * BM;
  const int N0 = nb * BN;

  // stage-1 wave roles: f sub-tile fw (16 f), token tiles ttbase..ttbase+3
  const int fw     = wave & 3;
  const int ttbase = (wave >> 2) * 4;
  // stage-2 wave roles: 2 waves along M (64 tokens = 4 tiles),
  //                     4 waves along N (32 cols  = 2 tiles)
  const int mw = wave >> 2;
  const int nw = wave & 3;

  float th[NQ];
  #pragma unroll
  for (int i = 0; i < NQ; ++i) th[i] = theta[i];

  union Frag  { v16h v; h16 e[16]; };
  union HOut  { v8h  v; h16 e[8];  };

  // ---- q^T B-fragments: B[k][tok] = cos(x[tok][k] + th[k]), k<8, pad K->32.
  // B f16 32x16 layout: lane=col, VGPR j (j<4) holds K = hi*8 + 2j, 2j+1, as for A.
  const float INV2PI = 0.15915494309189535f;
  Frag qf[4];
  #pragma unroll
  for (int j = 0; j < 4; ++j) {
    #pragma unroll
    for (int i = 0; i < 16; ++i) qf[j].e[i] = (h16)0.f;
    if (hi == 0) {                                    // K = 0..7 live here
      int tok = M0 + (ttbase + j) * 16 + r;
      const float* xr = x + (size_t)tok * EMBED;
      #pragma unroll
      for (int i = 0; i < NQ; ++i)
        qf[j].e[i] = (h16)__builtin_amdgcn_cosf((xr[i] + th[i]) * INV2PI);
    }
  }

  // ---- accumulators pre-loaded with b2 (per-column = per-lane constant) ----
  v8f acc[4][2];
  #pragma unroll
  for (int nt = 0; nt < 2; ++nt) {
    float bias = b2[N0 + (nw * 2 + nt) * 16 + r];
    #pragma unroll
    for (int mt = 0; mt < 4; ++mt)
      #pragma unroll
      for (int v = 0; v < 8; ++v) acc[mt][nt][v] = bias;
  }

  const v8h zeroh = {(h16)0, (h16)0, (h16)0, (h16)0,
                     (h16)0, (h16)0, (h16)0, (h16)0};

  // per-lane w2h base pointers for the two N tiles (bumped by KC each chunk)
  const h16* wbase[2];
  #pragma unroll
  for (int nt = 0; nt < 2; ++nt) {
    int e = N0 + (nw * 2 + nt) * 16 + r;
    wbase[nt] = w2h + (size_t)e * FFN + hi * 16;
  }

  int buf = 0;
  for (int kc = 0; kc < FFN; kc += KC) {
    h16* hbuf = hlds + buf * (BM * HSTR);

    // ---- issue stage-2 B-fragment loads EARLY (no LDS/barrier dependency) --
    // B: lane=col, contiguous K run of 16 at hi*16; two runs per KC chunk.
    Frag bfr[2][2];                                   // [ks][nt]
    #pragma unroll
    for (int ks = 0; ks < 2; ++ks)
      #pragma unroll
      for (int nt = 0; nt < 2; ++nt) {
        const h16* wp = wbase[nt] + ks * 32;
        *(v8h*)&bfr[ks][nt].e[0] = *(const v8h*)wp;        // global_load_b128
        *(v8h*)&bfr[ks][nt].e[8] = *(const v8h*)(wp + 8);  // global_load_b128
      }
    wbase[0] += KC;
    wbase[1] += KC;

    // ================= stage 1: h^T tile [KC x BM] -> LDS as h[tok][f] ======
    const int F0 = kc + fw * 16;
    // A-fragment (w1 rows, K padded 8->32): A layout lane=row,
    // VGPR0-3: K = hi*8 + 2j,2j+1 ; VGPR4-7: K = 16 + hi*8 + ...
    Frag af;
    #pragma unroll
    for (int i = 0; i < 16; ++i) af.e[i] = (h16)0.f;
    if (hi == 0) {                                    // K = 0..7 live here
      const float* wr = w1 + (size_t)(F0 + r) * NQ;
      #pragma unroll
      for (int i = 0; i < NQ; ++i) af.e[i] = (h16)wr[i];
    }
    // b1 as WMMA C operand: C[v] = b1[F0 + hi*8 + v] (same for every column)
    v8f cbias;
    {
      const float* b1p = b1 + F0 + hi * 8;
      #pragma unroll
      for (int v = 0; v < 8; ++v) cbias[v] = b1p[v];
    }

    #pragma unroll
    for (int j = 0; j < 4; ++j) {
      v8f d = __builtin_amdgcn_wmma_f32_16x16x32_f16(
          false, af.v, false, qf[j].v, (short)0, cbias, false, false);
      // D f32: col = token = (ttbase+j)*16 + r ; row v -> f = F0 + hi*8 + v
      HOut o;
      #pragma unroll
      for (int v = 0; v < 8; ++v) o.e[v] = (h16)d[v];  // v_cvt_pk_f16_f32 x4
      o.v = __builtin_elementwise_max(o.v, zeroh);     // v_pk_max_num_f16 x4
      int tloc = (ttbase + j) * 16 + r;
      *(v8h*)(hbuf + tloc * HSTR + fw * 16 + hi * 8) = o.v;  // ds_store_b128
    }
    __syncthreads();   // single barrier per chunk (double-buffered LDS)

    // ================= stage 2: acc += h_tile @ w2^T ========================
    #pragma unroll
    for (int ks = 0; ks < 2; ++ks) {
      const int k0 = ks * 32;
      v16h afr[4];
      #pragma unroll
      for (int mt = 0; mt < 4; ++mt) {
        int tloc = (mw * 4 + mt) * 16 + r;
        const h16* hp = hbuf + tloc * HSTR + k0 + hi * 16;
        Frag f;                                       // A: contiguous run of 16 at hi*16
        *(v8h*)&f.e[0] = *(const v8h*)hp;             // ds_load_b128
        *(v8h*)&f.e[8] = *(const v8h*)(hp + 8);       // ds_load_b128
        afr[mt] = f.v;
      }
      #pragma unroll
      for (int mt = 0; mt < 4; ++mt)
        #pragma unroll
        for (int nt = 0; nt < 2; ++nt)
          acc[mt][nt] = __builtin_amdgcn_wmma_f32_16x16x32_f16(
              false, afr[mt], false, bfr[ks][nt].v, (short)0, acc[mt][nt],
              false, false);
    }
    buf ^= 1;
  }

  // ================= epilogue: pure stores (b2 already in acc) ==============
  #pragma unroll
  for (int nt = 0; nt < 2; ++nt) {
    int e = N0 + (nw * 2 + nt) * 16 + r;              // col = lane
    #pragma unroll
    for (int mt = 0; mt < 4; ++mt) {
      int tokbase = M0 + (mw * 4 + mt) * 16 + hi * 8; // row = hi*8 + v
      float* op = out + (size_t)tokbase * EMBED + e;
      #pragma unroll
      for (int v = 0; v < 8; ++v)
        op[(size_t)v * EMBED] = acc[mt][nt][v];
    }
  }
}

// ---------------------------------------------------------------------------
extern "C" void kernel_launch(void* const* d_in, const int* in_sizes, int n_in,
                              void* d_out, int out_size, void* d_ws, size_t ws_size,
                              hipStream_t stream) {
  const float* x  = (const float*)d_in[0];
  const float* th = (const float*)d_in[1];
  const float* w1 = (const float*)d_in[2];
  const float* b1 = (const float*)d_in[3];
  const float* w2 = (const float*)d_in[4];
  const float* b2 = (const float*)d_in[5];
  float* out = (float*)d_out;

  // workspace: f16 copy of w2, EMBED*FFN*2 = 8 MiB (recomputed every call).
  h16* w2h = (h16*)d_ws;
  (void)ws_size; (void)in_sizes; (void)n_in; (void)out_size;

  w2_cvt_kernel<<<(EMBED * FFN) / (256 * 8), 256, 0, stream>>>(w2, w2h);

  dim3 grid((NTOK / BM) * (EMBED / BN));   // 128 * 8 = 1024 blocks
  ffq_fused_kernel<<<grid, 256, 0, stream>>>(x, th, w1, b1, w2h, b2, out);
}
